// DecoderLSTM_5549097746436
// MI455X (gfx1250) — matrix operand order repaired and checked by reading, not compiled, not run on hardware
//
#include <hip/hip_runtime.h>

typedef float  v8f    __attribute__((ext_vector_type(8)));
typedef __bf16 v16bf  __attribute__((ext_vector_type(16)));
typedef __bf16 bf16x8 __attribute__((ext_vector_type(8)));

#define NPED 8192
#define HDIM 256
#define EDIM 64
#define ROWS 64          // rows per block
#define AROW 584         // bf16 row stride: 64 emb | 256 h_even | 256 h_odd | 8 pad
#define CROW 260         // f32 c row stride (padded)
#define SEQ  30
#define LEAK 0.01f
#define EPSF 1e-5f

struct Smem {
  __bf16 A[ROWS * AROW];     // activations (bf16): emb + double-buffered h
  float  C[ROWS * CROW];     // cell state (f32)
  float  bias[1024];         // b_ih + b_hh
  float  ln2g[256], ln2b[256];
  float  posW[512];          // pos_W [2][256] row-major
  float  embW[128];          // emb_W [64][2]
  float  embB[64];
  float  rel[ROWS * 2];
  float  misc[8];            // ln1_g0, ln1_g1, ln1_b0, ln1_b1, pos_b0, pos_b1
};

__device__ __forceinline__ float sigmoidf_(float x) { return 1.0f / (1.0f + __expf(-x)); }

__device__ __forceinline__ v16bf combine16(bf16x8 lo, bf16x8 hi) {
  return __builtin_shufflevector(lo, hi, 0,1,2,3,4,5,6,7,8,9,10,11,12,13,14,15);
}

// 16-bit A-matrix 16x32 layout: lanes 0-15 = rows, K 0..7 in v0-3 and 16..23 in v4-7;
// lanes 16-31 same rows, K 8..15 / 24..31.  Two 16B LDS loads per lane.
__device__ __forceinline__ v16bf load_a_tile(const __bf16* p) {
  bf16x8 lo = *(const bf16x8*)p;
  bf16x8 hi = *(const bf16x8*)(p + 16);
  return combine16(lo, hi);
}

// 16-bit B-matrix 32x16 (KxN): lane n = column; lanes 0-15 hold K 0..7 / 16..23,
// lanes 16-31 hold K 8..15 / 24..31.  W stored [N][Kdim] row-major -> two 16B loads.
__device__ __forceinline__ v16bf load_b_tile(const __bf16* p) {
  bf16x8 lo = *(const bf16x8*)p;
  bf16x8 hi = *(const bf16x8*)(p + 16);
  return combine16(lo, hi);
}

// embed(rel): leaky_relu( LN1(rel) @ emb_W^T + emb_b ) -> bf16 into sA[:, 0:64]
__device__ __forceinline__ void embed_phase(Smem& sm, int tid) {
  int row = tid >> 2;
  int c0  = (tid & 3) * 16;
  float r0 = sm.rel[row * 2 + 0], r1 = sm.rel[row * 2 + 1];
  float d  = (r0 - r1) * 0.5f;                 // LN over 2 elems: closed form
  float rs = rsqrtf(d * d + EPSF);
  float x0 =  d * rs * sm.misc[0] + sm.misc[2];
  float x1 = -d * rs * sm.misc[1] + sm.misc[3];
#pragma unroll
  for (int cc = 0; cc < 16; ++cc) {
    int c = c0 + cc;
    float e = x0 * sm.embW[c * 2 + 0] + x1 * sm.embW[c * 2 + 1] + sm.embB[c];
    e = e > 0.0f ? e : LEAK * e;
    sm.A[row * AROW + c] = (__bf16)e;
  }
}

__global__ void prep_kernel(const float* __restrict__ Wih, const float* __restrict__ Whh,
                            const float* __restrict__ bih, const float* __restrict__ bhh,
                            __bf16* __restrict__ wihB, __bf16* __restrict__ whhB,
                            float* __restrict__ biasG) {
  int i = blockIdx.x * blockDim.x + threadIdx.x;
  if (i < 1024 * 64)  wihB[i] = (__bf16)Wih[i];
  if (i < 1024 * 256) whhB[i] = (__bf16)Whh[i];
  if (i < 1024)       biasG[i] = bih[i] + bhh[i];
}

__global__ __launch_bounds__(256)
void lstm_kernel(const float* __restrict__ last_pos_rel,
                 const float* __restrict__ h0, const float* __restrict__ c0,
                 const float* __restrict__ emb_W, const float* __restrict__ emb_b,
                 const float* __restrict__ ln1_g, const float* __restrict__ ln1_b,
                 const float* __restrict__ pos_W, const float* __restrict__ pos_b,
                 const float* __restrict__ ln2_g, const float* __restrict__ ln2_b,
                 const __bf16* __restrict__ wihB, const __bf16* __restrict__ whhB,
                 const float* __restrict__ biasG, float* __restrict__ out) {
  __shared__ Smem sm;
  const int tid = threadIdx.x;
  const int rowBase = blockIdx.x * ROWS;

  // ---- stage parameters into LDS ----
  for (int i = tid; i < 1024; i += 256) sm.bias[i] = biasG[i];
  for (int i = tid; i < 256;  i += 256) { sm.ln2g[i] = ln2_g[i]; sm.ln2b[i] = ln2_b[i]; }
  for (int i = tid; i < 512;  i += 256) sm.posW[i] = pos_W[i];
  for (int i = tid; i < 128;  i += 256) sm.embW[i] = emb_W[i];
  for (int i = tid; i < 64;   i += 256) sm.embB[i] = emb_b[i];
  if (tid == 0) {
    sm.misc[0] = ln1_g[0]; sm.misc[1] = ln1_g[1];
    sm.misc[2] = ln1_b[0]; sm.misc[3] = ln1_b[1];
    sm.misc[4] = pos_b[0]; sm.misc[5] = pos_b[1];
  }
  // ---- stage h0 (bf16, even buffer) and c0 (f32) ----
  for (int i = tid; i < ROWS * HDIM; i += 256) {
    int r = i >> 8, c = i & 255;
    sm.A[r * AROW + 64 + c] = (__bf16)h0[(size_t)(rowBase + r) * HDIM + c];
    sm.C[r * CROW + c]      = c0[(size_t)(rowBase + r) * HDIM + c];
  }
  for (int i = tid; i < ROWS * 2; i += 256) sm.rel[i] = last_pos_rel[(size_t)rowBase * 2 + i];
  __syncthreads();
  embed_phase(sm, tid);         // dec_in = embed(last_pos_rel)
  __syncthreads();

  const int wave = tid >> 5;
  const int lane = tid & 31;
  const int jt0  = wave * 2;    // this wave owns hidden col tiles jt0, jt0+1 (32 units)
  // Per-lane constant address parts
  const int aLaneRow = (lane & 15);          // A: row within tile
  const int aLaneK   = (lane >> 4) << 3;     // A: +8 elems for upper half-wave
  const int bLaneN   = (lane & 15);          // B: column within tile
  const int bLaneK   = (lane >> 4) << 3;     // B: +8 elems for upper half-wave

  for (int t = 0; t < SEQ; ++t) {
    const int hCur  = 64 + (t & 1) * 256;        // bf16 offset of current h
    const int hNext = 64 + ((t + 1) & 1) * 256;  // where new h goes

    // ================= Phase A: gates GEMM + cell update =================
#pragma unroll 1
    for (int mp = 0; mp < 2; ++mp) {
#pragma unroll 1
      for (int jl = 0; jl < 2; ++jl) {
        const int nt = (jt0 + jl) * 16;                       // hidden-unit col base
        // A row base pointers for the two m-tiles of this pair
        const __bf16* aRow0 = sm.A + ((mp * 2 + 0) * 16 + aLaneRow) * AROW + aLaneK;
        const __bf16* aRow1 = sm.A + ((mp * 2 + 1) * 16 + aLaneRow) * AROW + aLaneK;
        // B row base pointers per gate (W stored [4H][Kdim] row-major)
        const __bf16* bih_[4];
        const __bf16* bhh_[4];
#pragma unroll
        for (int g = 0; g < 4; ++g) {
          const int n = g * 256 + nt + bLaneN;
          bih_[g] = wihB + n * EDIM + bLaneK;
          bhh_[g] = whhB + n * HDIM + bLaneK;
        }

        v8f acc[2][4];  // [m-tile local][gate i,f,g,o]
#pragma unroll
        for (int g = 0; g < 4; ++g) {
          const float bv = sm.bias[g * 256 + nt + (lane & 15)];
#pragma unroll
          for (int r = 0; r < 8; ++r) { acc[0][g][r] = bv; acc[1][g][r] = bv; }
        }

        // ---- K over embedding input (W_ih), 2 k-tiles ----
#pragma unroll
        for (int kc = 0; kc < 2; ++kc) {
          v16bf a0 = load_a_tile(aRow0 + kc * 32);
          v16bf a1 = load_a_tile(aRow1 + kc * 32);
#pragma unroll
          for (int g = 0; g < 4; ++g) {
            v16bf b = load_b_tile(bih_[g] + kc * 32);
            acc[0][g] = __builtin_amdgcn_wmma_f32_16x16x32_bf16(
                false, a0, false, b, (short)0, acc[0][g], false, false);
            acc[1][g] = __builtin_amdgcn_wmma_f32_16x16x32_bf16(
                false, a1, false, b, (short)0, acc[1][g], false, false);
          }
        }
        // ---- K over hidden state (W_hh), 8 k-tiles ----
#pragma unroll 2
        for (int kc = 0; kc < 8; ++kc) {
          v16bf a0 = load_a_tile(aRow0 + hCur + kc * 32);
          v16bf a1 = load_a_tile(aRow1 + hCur + kc * 32);
#pragma unroll
          for (int g = 0; g < 4; ++g) {
            v16bf b = load_b_tile(bhh_[g] + kc * 32);
            acc[0][g] = __builtin_amdgcn_wmma_f32_16x16x32_bf16(
                false, a0, false, b, (short)0, acc[0][g], false, false);
            acc[1][g] = __builtin_amdgcn_wmma_f32_16x16x32_bf16(
                false, a1, false, b, (short)0, acc[1][g], false, false);
          }
        }

        // ---- cell update: 32 rows x 16 cols just produced ----
        const int j = nt + (lane & 15);
#pragma unroll
        for (int ml = 0; ml < 2; ++ml) {
          const int rbase = (mp * 2 + ml) * 16 + ((lane >> 4) << 3);
#pragma unroll
          for (int r = 0; r < 8; ++r) {          // C layout: VGPR r -> row r / r+8
            const int row = rbase + r;
            float iv = sigmoidf_(acc[ml][0][r]);
            float fv = sigmoidf_(acc[ml][1][r]);
            float gv = tanhf(acc[ml][2][r]);
            float ov = sigmoidf_(acc[ml][3][r]);
            float cn = fv * sm.C[row * CROW + j] + iv * gv;
            sm.C[row * CROW + j] = cn;
            sm.A[row * AROW + hNext + j] = (__bf16)(ov * tanhf(cn));
          }
        }
      }
    }
    __syncthreads();

    // ================= Phase B: rel = sigmoid(LN2(h) @ pos_W^T + pos_b) =================
    if (tid < ROWS) {
      const __bf16* hp = sm.A + tid * AROW + hNext;
      float s = 0.f, ss = 0.f;
      for (int k = 0; k < 256; ++k) { float v = (float)hp[k]; s += v; ss += v * v; }
      float mu  = s * (1.0f / 256.0f);
      float var = ss * (1.0f / 256.0f) - mu * mu;
      float rsv = rsqrtf(var + EPSF);
      float a0 = sm.misc[4], a1 = sm.misc[5];
      for (int k = 0; k < 256; ++k) {
        float hn = ((float)hp[k] - mu) * rsv * sm.ln2g[k] + sm.ln2b[k];
        a0 += hn * sm.posW[k];
        a1 += hn * sm.posW[256 + k];
      }
      float r0 = sigmoidf_(a0), r1 = sigmoidf_(a1);
      sm.rel[tid * 2 + 0] = r0;
      sm.rel[tid * 2 + 1] = r1;
      float2* op = (float2*)(out + ((size_t)t * NPED + rowBase + tid) * 2);
      *op = make_float2(r0, r1);
    }
    __syncthreads();

    // ================= Phase C: dec_in = embed(rel) =================
    embed_phase(sm, tid);
    __syncthreads();
  }
}

extern "C" void kernel_launch(void* const* d_in, const int* in_sizes, int n_in,
                              void* d_out, int out_size, void* d_ws, size_t ws_size,
                              hipStream_t stream) {
  const float* last_pos_rel = (const float*)d_in[1];
  const float* h0    = (const float*)d_in[2];
  const float* c0    = (const float*)d_in[3];
  const float* W_ih  = (const float*)d_in[4];
  const float* W_hh  = (const float*)d_in[5];
  const float* b_ih  = (const float*)d_in[6];
  const float* b_hh  = (const float*)d_in[7];
  const float* emb_W = (const float*)d_in[8];
  const float* emb_b = (const float*)d_in[9];
  const float* ln1_g = (const float*)d_in[10];
  const float* ln1_b = (const float*)d_in[11];
  const float* pos_W = (const float*)d_in[12];
  const float* pos_b = (const float*)d_in[13];
  const float* ln2_g = (const float*)d_in[14];
  const float* ln2_b = (const float*)d_in[15];

  __bf16* wihB  = (__bf16*)d_ws;                                   // 1024*64 bf16
  __bf16* whhB  = (__bf16*)((char*)d_ws + 131072);                 // 1024*256 bf16
  float*  biasG = (float*)((char*)d_ws + 131072 + 524288);         // 1024 f32

  prep_kernel<<<1024, 256, 0, stream>>>(W_ih, W_hh, b_ih, b_hh, wihB, whhB, biasG);
  lstm_kernel<<<NPED / ROWS, 256, 0, stream>>>(
      last_pos_rel, h0, c0, emb_W, emb_b, ln1_g, ln1_b, pos_W, pos_b,
      ln2_g, ln2_b, wihB, whhB, biasG, (float*)d_out);
}
